// CrystalGraphConvNet_55585466745327
// MI455X (gfx1250) — compile-verified, matrix operand order checked by reading, not run on hardware
//
#include <hip/hip_runtime.h>

typedef __attribute__((ext_vector_type(16))) _Float16 v16h;
typedef __attribute__((ext_vector_type(8)))  float    v8f;

#define N_NODES   100000
#define N_EDGES   1600000
#define N_GRAPHS  256
#define NODE_DIM  64
#define EDGE_DIM  64
#define HIDDEN    128
#define N_CONV    3
#define LN_EPS    1e-5f

// Gaussian expansion constants: centers = linspace(1,6,64); step = 5/64
#define G_CMIN    1.0f
#define G_CSTEP   (5.0f / 63.0f)
#define G_INVSTEP (64.0f / 5.0f)

// ---- workspace layout (bytes) ----
#define X_OFF     ((size_t)0)
#define MSG_OFF   ((size_t)N_NODES * NODE_DIM * 4)                    // 25,600,000
#define H1_OFF    ((size_t)2 * N_NODES * NODE_DIM * 4)                // 51,200,000
#define H2_OFF    ((size_t)0)
#define FRAG_OFF  ((size_t)102400000)
#define FRAG_HALVES 98304                    // 6*12288 (conv) + 8192 (W1) + 16384 (W2)
#define FRAG_W1   73728
#define FRAG_W2   81920
#define SUMS_OFF  (FRAG_OFF + (size_t)FRAG_HALVES * 2)
#define CNT_OFF   (SUMS_OFF + (size_t)N_GRAPHS * 4)

__device__ __forceinline__ float sigmoidf_(float x) { return 1.0f / (1.0f + __expf(-x)); }
__device__ __forceinline__ float softplusf_(float x) { return (x > 20.0f) ? x : log1pf(__expf(x)); }

// Fragment element -> K index inside a 32-wide K chunk (same formula for 16-bit A and B):
// lane l: row/col = l&15 ; element e: K = (l>>4)*8 + (e&7) + ((e>>3)<<4)

// ---------------- weight fragment prep: fp32 [K,N] -> f16 WMMA B fragments ----------------
__global__ void prep_frags_kernel(const float* __restrict__ Wf, const float* __restrict__ Ws,
                                  const float* __restrict__ W1, const float* __restrict__ W2,
                                  _Float16* __restrict__ frags) {
  int t = blockIdx.x * blockDim.x + threadIdx.x;
  if (t >= FRAG_HALVES) return;
  int e    = t & 15;
  int lane = (t >> 4) & 31;
  int kin  = (lane >> 4) * 8 + (e & 7) + ((e >> 3) << 4);
  int nloc = lane & 15;
  float val;
  if (t < FRAG_W1) {                      // conv weights: 6 matrices of [192,64]
    int r      = t >> 9;                  // matIdx*24 + kc*4 + nt
    int nt     = r & 3;
    int kcm    = r >> 2;                  // matIdx*6 + kc
    int kc     = kcm % 6;
    int matIdx = kcm / 6;                 // layer*2 + (0=Wf,1=Ws)
    const float* W = (matIdx & 1) ? Ws : Wf;
    int layer  = matIdx >> 1;
    val = W[((size_t)layer * 192 + (kc * 32 + kin)) * 64 + nt * 16 + nloc];
  } else if (t < FRAG_W2) {               // W1: [64,128], KC=2, NT=8
    int r  = (t - FRAG_W1) >> 9;          // kc*8 + nt
    int nt = r & 7, kc = r >> 3;
    val = W1[(size_t)(kc * 32 + kin) * 128 + nt * 16 + nloc];
  } else {                                // W2: [128,128], KC=4, NT=8
    int r  = (t - FRAG_W2) >> 9;
    int nt = r & 7, kc = r >> 3;
    val = W2[(size_t)(kc * 32 + kin) * 128 + nt * 16 + nloc];
  }
  frags[t] = (_Float16)val;
}

// ---------------- misc small kernels ----------------
__global__ void zero_f32_kernel(float* __restrict__ p, int n) {
  int t = blockIdx.x * blockDim.x + threadIdx.x;
  if (t < n) p[t] = 0.0f;
}

__global__ void embed_kernel(const int* __restrict__ numbers, const float* __restrict__ table,
                             float* __restrict__ x) {
  int t = blockIdx.x * blockDim.x + threadIdx.x;
  if (t >= N_NODES * NODE_DIM) return;
  int n = t >> 6, d = t & 63;
  x[t] = table[((size_t)numbers[n] << 6) + d];
}

// ---------------- edge conv: z[16,192] @ {Wf,Ws}[192,64] via WMMA, gate, atomic scatter ----------------
// fragFS = 24576 halves: Wf fragments (12288) followed by Ws fragments (12288), staged in LDS.
__global__ void edge_conv_kernel(const float* __restrict__ x,
                                 const int* __restrict__ src, const int* __restrict__ tgt,
                                 const float* __restrict__ elen,
                                 const _Float16* __restrict__ fragFS,
                                 const float* __restrict__ bf, const float* __restrict__ bs,
                                 float* __restrict__ msg, int ntiles) {
  __shared__ alignas(16) _Float16 lds[24576];
  for (int i = threadIdx.x; i < 3072; i += blockDim.x)
    ((uint4*)lds)[i] = ((const uint4*)fragFS)[i];
  __syncthreads();

  const int lane = threadIdx.x & 31;
  const int m    = lane & 15;
  const int half = lane >> 4;
  const int wid  = (blockIdx.x * blockDim.x + threadIdx.x) >> 5;
  const int nw   = (gridDim.x * blockDim.x) >> 5;
  const v8f vzero = {};
  int bofs = 0;   // always 0; laundered per-iteration to block LICM hoist of B frags

  for (int tile = wid; tile < ntiles; tile += nw) {
    asm volatile("" : "+v"(bofs));
    const int e0 = tile * 16 + m;
    const int si = src[e0];
    const int ti = tgt[e0];
    const float d = elen[e0];

    // Build A fragments: z = [x[src] | x[tgt] | gaussian(d)] in f16
    v16h a[6];
#pragma unroll
    for (int kc = 0; kc < 6; ++kc) {
#pragma unroll
      for (int g = 0; g < 2; ++g) {
        const int bK = kc * 32 + half * 8 + g * 16;   // run of 8 consecutive K, segment-uniform
        if (kc < 4) {
          const float* p = (kc < 2) ? (x + (size_t)si * NODE_DIM + bK)
                                    : (x + (size_t)ti * NODE_DIM + (bK - 64));
          float4 p0 = *(const float4*)p, p1 = *(const float4*)(p + 4);
          a[kc][g * 8 + 0] = (_Float16)p0.x; a[kc][g * 8 + 1] = (_Float16)p0.y;
          a[kc][g * 8 + 2] = (_Float16)p0.z; a[kc][g * 8 + 3] = (_Float16)p0.w;
          a[kc][g * 8 + 4] = (_Float16)p1.x; a[kc][g * 8 + 5] = (_Float16)p1.y;
          a[kc][g * 8 + 6] = (_Float16)p1.z; a[kc][g * 8 + 7] = (_Float16)p1.w;
        } else {
#pragma unroll
          for (int j = 0; j < 8; ++j) {
            float c = G_CMIN + (float)(bK - 128 + j) * G_CSTEP;
            float u = (d - c) * G_INVSTEP;
            a[kc][g * 8 + j] = (_Float16)__expf(-0.5f * u * u);
          }
        }
      }
    }

    v8f af[4], as_[4];
#pragma unroll
    for (int nt = 0; nt < 4; ++nt) { af[nt] = vzero; as_[nt] = vzero; }

#pragma unroll
    for (int kc = 0; kc < 6; ++kc) {
#pragma unroll
      for (int nt = 0; nt < 4; ++nt) {
        const int fo = bofs + (((kc * 4 + nt) * 32 + lane) << 4);
        v16h bF = *(const v16h*)(lds + fo);
        af[nt] = __builtin_amdgcn_wmma_f32_16x16x32_f16(false, a[kc], false, bF,
                                                        (short)0, af[nt], false, false);
        v16h bS = *(const v16h*)(lds + 12288 + fo);
        as_[nt] = __builtin_amdgcn_wmma_f32_16x16x32_f16(false, a[kc], false, bS,
                                                         (short)0, as_[nt], false, false);
      }
    }

    float bfv[4], bsv[4];
#pragma unroll
    for (int nt = 0; nt < 4; ++nt) { bfv[nt] = bf[nt * 16 + m]; bsv[nt] = bs[nt * 16 + m]; }

    // C layout: element r -> row M = r + half*8, col N = nt*16 + m
#pragma unroll
    for (int r = 0; r < 8; ++r) {
      const int srcm = __shfl(si, r + half * 8, 32);   // src of edge tile*16 + M
#pragma unroll
      for (int nt = 0; nt < 4; ++nt) {
        float gt = sigmoidf_(af[nt][r] + bfv[nt]) * softplusf_(as_[nt][r] + bsv[nt]);
        atomicAdd(msg + (size_t)srcm * NODE_DIM + nt * 16 + m, gt);
      }
    }
  }
}

// ---------------- conv LN + residual: x += LN(msg) ----------------
__global__ void ln_residual_kernel(float* __restrict__ x, const float* __restrict__ msg,
                                   const float* __restrict__ g, const float* __restrict__ b) {
  int n = blockIdx.x * blockDim.x + threadIdx.x;
  if (n >= N_NODES) return;
  const float* mp = msg + (size_t)n * NODE_DIM;
  float s = 0.0f, ss = 0.0f;
#pragma unroll
  for (int j = 0; j < 16; ++j) {
    float4 v = *(const float4*)(mp + j * 4);
    s  += v.x + v.y + v.z + v.w;
    ss += v.x * v.x + v.y * v.y + v.z * v.z + v.w * v.w;
  }
  float mu = s * (1.0f / 64.0f);
  float var = ss * (1.0f / 64.0f) - mu * mu;
  float rstd = rsqrtf(var + LN_EPS);
  float* xp = x + (size_t)n * NODE_DIM;
#pragma unroll
  for (int j = 0; j < 16; ++j) {
    float4 v  = *(const float4*)(mp + j * 4);
    float4 xo = *(const float4*)(xp + j * 4);
    float4 gv = *(const float4*)(g + j * 4);
    float4 bv = *(const float4*)(b + j * 4);
    xo.x += (v.x - mu) * rstd * gv.x + bv.x;
    xo.y += (v.y - mu) * rstd * gv.y + bv.y;
    xo.z += (v.z - mu) * rstd * gv.z + bv.z;
    xo.w += (v.w - mu) * rstd * gv.w + bv.w;
    *(float4*)(xp + j * 4) = xo;
  }
}

// ---------------- generic 16-row WMMA GEMM: out[N,NT*16] = in[N,KC*32] @ frag + bias ----------------
template <int KC, int NT>
__global__ void gemm16_kernel(const float* __restrict__ in, const _Float16* __restrict__ frag,
                              const float* __restrict__ bias, float* __restrict__ out, int ntiles) {
  constexpr int K  = KC * 32;
  constexpr int NO = NT * 16;
  constexpr int NFRAG_H = KC * NT * 512;           // halves
  __shared__ alignas(16) _Float16 lds[NFRAG_H];
  for (int i = threadIdx.x; i < NFRAG_H / 8; i += blockDim.x)
    ((uint4*)lds)[i] = ((const uint4*)frag)[i];
  __syncthreads();

  const int lane = threadIdx.x & 31;
  const int m    = lane & 15;
  const int half = lane >> 4;
  const int wid  = (blockIdx.x * blockDim.x + threadIdx.x) >> 5;
  const int nw   = (gridDim.x * blockDim.x) >> 5;
  const v8f vzero = {};
  int bofs = 0;   // laundered to block LICM hoist of B fragments

  for (int tile = wid; tile < ntiles; tile += nw) {
    asm volatile("" : "+v"(bofs));
    const float* row = in + (size_t)(tile * 16 + m) * K;
    v16h a[KC];
#pragma unroll
    for (int kc = 0; kc < KC; ++kc) {
#pragma unroll
      for (int g = 0; g < 2; ++g) {
        const float* p = row + kc * 32 + half * 8 + g * 16;
        float4 p0 = *(const float4*)p, p1 = *(const float4*)(p + 4);
        a[kc][g * 8 + 0] = (_Float16)p0.x; a[kc][g * 8 + 1] = (_Float16)p0.y;
        a[kc][g * 8 + 2] = (_Float16)p0.z; a[kc][g * 8 + 3] = (_Float16)p0.w;
        a[kc][g * 8 + 4] = (_Float16)p1.x; a[kc][g * 8 + 5] = (_Float16)p1.y;
        a[kc][g * 8 + 6] = (_Float16)p1.z; a[kc][g * 8 + 7] = (_Float16)p1.w;
      }
    }
    v8f acc[NT];
#pragma unroll
    for (int nt = 0; nt < NT; ++nt) acc[nt] = vzero;
#pragma unroll
    for (int kc = 0; kc < KC; ++kc) {
#pragma unroll
      for (int nt = 0; nt < NT; ++nt) {
        v16h bF = *(const v16h*)(lds + bofs + (((kc * NT + nt) * 32 + lane) << 4));
        acc[nt] = __builtin_amdgcn_wmma_f32_16x16x32_f16(false, a[kc], false, bF,
                                                         (short)0, acc[nt], false, false);
      }
    }
#pragma unroll
    for (int nt = 0; nt < NT; ++nt) {
      float bv = bias[nt * 16 + m];
#pragma unroll
      for (int r = 0; r < 8; ++r) {
        out[(size_t)(tile * 16 + r + half * 8) * NO + nt * 16 + m] = acc[nt][r] + bv;
      }
    }
  }
}

// ---------------- in-place LN + softplus over 128 dims ----------------
__global__ void ln_softplus_kernel(float* __restrict__ h, const float* __restrict__ g,
                                   const float* __restrict__ b) {
  int n = blockIdx.x * blockDim.x + threadIdx.x;
  if (n >= N_NODES) return;
  float* hp = h + (size_t)n * HIDDEN;
  float s = 0.0f, ss = 0.0f;
#pragma unroll 8
  for (int j = 0; j < 32; ++j) {
    float4 v = *(const float4*)(hp + j * 4);
    s  += v.x + v.y + v.z + v.w;
    ss += v.x * v.x + v.y * v.y + v.z * v.z + v.w * v.w;
  }
  float mu = s * (1.0f / 128.0f);
  float var = ss * (1.0f / 128.0f) - mu * mu;
  float rstd = rsqrtf(var + LN_EPS);
#pragma unroll 8
  for (int j = 0; j < 32; ++j) {
    float4 v  = *(const float4*)(hp + j * 4);
    float4 gv = *(const float4*)(g + j * 4);
    float4 bv = *(const float4*)(b + j * 4);
    v.x = softplusf_((v.x - mu) * rstd * gv.x + bv.x);
    v.y = softplusf_((v.y - mu) * rstd * gv.y + bv.y);
    v.z = softplusf_((v.z - mu) * rstd * gv.z + bv.z);
    v.w = softplusf_((v.w - mu) * rstd * gv.w + bv.w);
    *(float4*)(hp + j * 4) = v;
  }
}

// ---------------- per-node energy + per-graph accumulation ----------------
__global__ void out_accum_kernel(const float* __restrict__ h, const float* __restrict__ Wo,
                                 const float* __restrict__ bo, const int* __restrict__ batch,
                                 float* __restrict__ sums, float* __restrict__ cnt) {
  int n = blockIdx.x * blockDim.x + threadIdx.x;
  if (n >= N_NODES) return;
  const float* hp = h + (size_t)n * HIDDEN;
  float acc = 0.0f;
#pragma unroll 8
  for (int j = 0; j < 32; ++j) {
    float4 hv = *(const float4*)(hp + j * 4);
    float4 wv = *(const float4*)(Wo + j * 4);
    acc += hv.x * wv.x + hv.y * wv.y + hv.z * wv.z + hv.w * wv.w;
  }
  acc += bo[0];
  int gidx = batch[n];
  atomicAdd(&sums[gidx], acc);
  atomicAdd(&cnt[gidx], 1.0f);
}

__global__ void finalize_kernel(const float* __restrict__ sums, const float* __restrict__ cnt,
                                float* __restrict__ out) {
  int g = threadIdx.x;
  if (g < N_GRAPHS) out[g] = sums[g] / fmaxf(cnt[g], 1.0f);
}

// ---------------- host orchestration ----------------
extern "C" void kernel_launch(void* const* d_in, const int* in_sizes, int n_in,
                              void* d_out, int out_size, void* d_ws, size_t ws_size,
                              hipStream_t stream) {
  (void)in_sizes; (void)n_in; (void)out_size; (void)ws_size;
  const int*   numbers = (const int*)d_in[0];
  const int*   eidx    = (const int*)d_in[1];
  const int*   src     = eidx;
  const int*   tgt     = eidx + N_EDGES;
  const float* elen    = (const float*)d_in[2];
  const int*   batch   = (const int*)d_in[3];
  const float* emb     = (const float*)d_in[4];
  const float* Wf      = (const float*)d_in[5];
  const float* bf      = (const float*)d_in[6];
  const float* Ws      = (const float*)d_in[7];
  const float* bs      = (const float*)d_in[8];
  const float* lng     = (const float*)d_in[9];
  const float* lnb     = (const float*)d_in[10];
  const float* W1      = (const float*)d_in[11];
  const float* b1      = (const float*)d_in[12];
  const float* g1      = (const float*)d_in[13];
  const float* bt1     = (const float*)d_in[14];
  const float* W2      = (const float*)d_in[15];
  const float* b2      = (const float*)d_in[16];
  const float* g2      = (const float*)d_in[17];
  const float* bt2     = (const float*)d_in[18];
  const float* Wo      = (const float*)d_in[19];
  const float* bo      = (const float*)d_in[20];
  float* out = (float*)d_out;

  char* ws = (char*)d_ws;
  float*     x     = (float*)(ws + X_OFF);
  float*     msg   = (float*)(ws + MSG_OFF);
  float*     h1    = (float*)(ws + H1_OFF);
  float*     h2    = (float*)(ws + H2_OFF);          // aliases dead x/msg region
  _Float16*  frags = (_Float16*)(ws + FRAG_OFF);
  float*     sums  = (float*)(ws + SUMS_OFF);
  float*     cnt   = (float*)(ws + CNT_OFF);

  // 1. weight fragments (f16, WMMA B layout)
  prep_frags_kernel<<<(FRAG_HALVES + 255) / 256, 256, 0, stream>>>(Wf, Ws, W1, W2, frags);

  // 2. node embedding
  embed_kernel<<<(N_NODES * NODE_DIM + 255) / 256, 256, 0, stream>>>(numbers, emb, x);

  // 3. conv layers
  for (int i = 0; i < N_CONV; ++i) {
    zero_f32_kernel<<<(N_NODES * NODE_DIM + 255) / 256, 256, 0, stream>>>(msg, N_NODES * NODE_DIM);
    edge_conv_kernel<<<2048, 256, 0, stream>>>(
        x, src, tgt, elen,
        frags + (size_t)i * 24576,               // Wf frags + Ws frags, contiguous
        bf + i * NODE_DIM, bs + i * NODE_DIM, msg, N_EDGES / 16);
    ln_residual_kernel<<<(N_NODES + 255) / 256, 256, 0, stream>>>(x, msg, lng + i * NODE_DIM,
                                                                  lnb + i * NODE_DIM);
  }

  // 4. output MLP
  gemm16_kernel<2, 8><<<1024, 256, 0, stream>>>(x, frags + FRAG_W1, b1, h1, N_NODES / 16);
  ln_softplus_kernel<<<(N_NODES + 255) / 256, 256, 0, stream>>>(h1, g1, bt1);
  gemm16_kernel<4, 8><<<1024, 256, 0, stream>>>(h1, frags + FRAG_W2, b2, h2, N_NODES / 16);
  ln_softplus_kernel<<<(N_NODES + 255) / 256, 256, 0, stream>>>(h2, g2, bt2);

  // 5. per-graph mean energy
  zero_f32_kernel<<<2, 256, 0, stream>>>(sums, 2 * N_GRAPHS);  // sums + cnt contiguous
  out_accum_kernel<<<(N_NODES + 255) / 256, 256, 0, stream>>>(h2, Wo, bo, batch, sums, cnt);
  finalize_kernel<<<1, 256, 0, stream>>>(sums, cnt, out);
}
